// EvoformerIteration_12446815224385
// MI455X (gfx1250) — matrix operand order repaired and checked by reading, not compiled, not run on hardware
//
#include <hip/hip_runtime.h>
#include <math.h>

// ---------------------------------------------------------------------------
// CDNA5 (gfx1250) Evoformer iteration, wave32 + V_WMMA_F32_16X16X32_BF16.
// ---------------------------------------------------------------------------

typedef __bf16 bf16;
typedef __attribute__((ext_vector_type(16))) __bf16 v16bf;
typedef __attribute__((ext_vector_type(8)))  float  v8f;

#define S_DIM  128
#define N_DIM  256
#define DMSA   256
#define DPAIR  128
#define HMSA   8
#define HPAIR  4
#define BIGF   8388608LL   // S*N*DMSA == N*N*DPAIR (all big bufs equal)

#define WAIT_DS() asm volatile("s_wait_dscnt 0" ::: "memory")

__device__ __forceinline__ v8f wmma_bf16(v16bf a, v16bf b, v8f c) {
  return __builtin_amdgcn_wmma_f32_16x16x32_bf16(false, a, false, b, (short)0, c,
                                                 false, false);
}

// A-frag: 16(M) x 32(K) bf16 from LDS, row-major with leading dim ld.
// lane L holds row M=L&15; K = 16*(L>>4) + e   (matches B-frag order)
__device__ __forceinline__ v16bf load_a_frag(const bf16* p, int ld) {
  const int lane = threadIdx.x & 31;
  const bf16* row = p + (lane & 15) * ld;
  const int kb = (lane >> 4) * 16;
  v16bf f;
#pragma unroll
  for (int e = 0; e < 16; ++e) f[e] = row[kb + e];
  return f;
}

// B-frag: 32(K) x 16(N) bf16; LDS holds B TRANSPOSED as [n][k] (ld = k-pitch)
// lane L holds col N=L&15, K = 16*(L>>4)+e  -> 16 contiguous bf16 per lane.
__device__ __forceinline__ v16bf load_b_frag_t(const bf16* pT, int ld) {
  const int lane = threadIdx.x & 31;
  const bf16* col = pT + (lane & 15) * ld + ((lane >> 4) << 4);
  v16bf f;
#pragma unroll
  for (int e = 0; e < 16; ++e) f[e] = col[e];
  return f;
}

// B-frag from a GLOBAL pre-transposed bf16 weight [n][k] (ldk = K pitch):
// per lane 16 contiguous bf16 (32B) -> vectorized global loads.
__device__ __forceinline__ v16bf load_b_frag_gt(const bf16* __restrict__ Wt,
                                                int ldk, int k0, int n) {
  const bf16* p =
      Wt + (long long)n * ldk + k0 + (((threadIdx.x & 31) >> 4) << 4);
  v16bf f;
#pragma unroll
  for (int e = 0; e < 16; ++e) f[e] = p[e];
  return f;
}

// ---------------------------------------------------------------------------
// One-shot weight transpose f32[K][N] -> bf16[N][K]  (N is a power of two)
// ---------------------------------------------------------------------------
__global__ __launch_bounds__(256) void transpose_bf16_kernel(
    const float* __restrict__ Wsrc, bf16* __restrict__ Wt, int K, int nsh) {
  const int N = 1 << nsh;
  const long long total = (long long)K << nsh;
  for (long long idx = (long long)blockIdx.x * 256 + threadIdx.x; idx < total;
       idx += (long long)gridDim.x * 256) {
    const int k = (int)(idx >> nsh);
    const int n = (int)(idx & (N - 1));
    Wt[(long long)n * K + k] = (bf16)Wsrc[idx];
  }
}

// ---------------------------------------------------------------------------
// LayerNorm: one wave per row; composite row map off = (r/L)*sb + (r%L)*sl
// ---------------------------------------------------------------------------
__global__ __launch_bounds__(256) void ln_kernel(
    const float* __restrict__ X, long long sb, long long sl, int L,
    const float* __restrict__ g, const float* __restrict__ b,
    float* __restrict__ Y, int M, int D) {
  const int wave = threadIdx.x >> 5, lane = threadIdx.x & 31;
  const int row = blockIdx.x * 8 + wave;
  if (row >= M) return;
  const long long ro = (sb == 0) ? (long long)row * sl
                                 : (long long)(row / L) * sb +
                                       (long long)(row % L) * sl;
  float s = 0.f, ss = 0.f;
  for (int d = lane; d < D; d += 32) {
    float v = X[ro + d];
    s += v; ss += v * v;
  }
#pragma unroll
  for (int off = 16; off >= 1; off >>= 1) {
    s += __shfl_xor(s, off, 32);
    ss += __shfl_xor(ss, off, 32);
  }
  const float mu = s / D;
  float var = ss / D - mu * mu;
  var = var < 0.f ? 0.f : var;
  const float inv = rsqrtf(var + 1e-5f);
  for (int d = lane; d < D; d += 32)
    Y[(long long)row * D + d] = (X[ro + d] - mu) * inv * g[d] + b[d];
}

// ---------------------------------------------------------------------------
// Generic GEMM: C = ep( A[M,K] @ W[K,N] + bias ), 64x64 block tile, 8 waves,
// 2 wmma / wave / k-chunk, double-buffered LDS staging (global loads for the
// next k-chunk retire under the barrier + wmmas of the current chunk).
// Requires: K % 32 == 0 and M % 64 == 0 (true for every call site).
// ep: act(0 none,1 relu,2 sigmoid) * mul[m,N] * rowscale[m] (+C if addC)
// ---------------------------------------------------------------------------
__global__ __launch_bounds__(256) void gemm_kernel(
    const float* __restrict__ A, long long asb, long long asl, int aL,
    const float* __restrict__ W, const float* __restrict__ bias,
    float* __restrict__ C, long long csb, long long csl, int cL,
    const float* __restrict__ mulbuf, const float* __restrict__ rowscale,
    int M, int K, int N, int act, int addC) {
  __shared__ bf16 aS[2][64][32];
  __shared__ bf16 wS[2][64][32];   // transposed [n][k]
  const int tid = threadIdx.x;
  const int lane = tid & 31, wave = tid >> 5;
  const int m0 = blockIdx.x * 64, n0 = blockIdx.y * 64;
  const int mrow = (wave >> 1) * 16, ncol = (wave & 1) * 32;
  // thread-constant staging coordinates (k-invariant; computed once)
  const int kq = (tid & 7) * 4;   // float4 slot within the 32-wide k-chunk
  const int ar0 = tid >> 3;       // 0..31 -> rows ar0, ar0+32
  const int wn = tid & 63;        // W stage column
  const int wk = tid >> 6;        // W stage k base (0..3), rows wk+4*it
  const bool wok = (n0 + wn) < N;
  const int mA0 = m0 + ar0, mA1 = mA0 + 32;
  const long long roA0 =
      (asb == 0) ? (long long)mA0 * asl
                 : (long long)(mA0 / aL) * asb + (long long)(mA0 % aL) * asl;
  const long long roA1 =
      (asb == 0) ? (long long)mA1 * asl
                 : (long long)(mA1 / aL) * asb + (long long)(mA1 % aL) * asl;
  const float* wp = W + n0 + wn;

  // prologue: load chunk 0 into registers
  float4 a0 = *(const float4*)(A + roA0 + kq);
  float4 a1 = *(const float4*)(A + roA1 + kq);
  float wv[8];
#pragma unroll
  for (int it = 0; it < 8; ++it)
    wv[it] = wok ? wp[(long long)(wk + it * 4) * N] : 0.f;

  v8f accs[2] = {{}, {}};
  int buf = 0;
  for (int kc = 0; kc < K; kc += 32) {
    // ---- stage current chunk into LDS[buf] ----
    bf16* arow0 = &aS[buf][ar0][kq];
    arow0[0] = (bf16)a0.x; arow0[1] = (bf16)a0.y;
    arow0[2] = (bf16)a0.z; arow0[3] = (bf16)a0.w;
    bf16* arow1 = &aS[buf][ar0 + 32][kq];
    arow1[0] = (bf16)a1.x; arow1[1] = (bf16)a1.y;
    arow1[2] = (bf16)a1.z; arow1[3] = (bf16)a1.w;
#pragma unroll
    for (int it = 0; it < 8; ++it) wS[buf][wn][wk + it * 4] = (bf16)wv[it];
    // ---- issue next chunk's global loads (overlap barrier + wmma) ----
    if (kc + 32 < K) {
      a0 = *(const float4*)(A + roA0 + kc + 32 + kq);
      a1 = *(const float4*)(A + roA1 + kc + 32 + kq);
      const float* wpn = wp + (long long)(kc + 32) * N;
#pragma unroll
      for (int it = 0; it < 8; ++it)
        wv[it] = wok ? wpn[(long long)(wk + it * 4) * N] : 0.f;
    }
    if (kc + 64 < K)  // global_prefetch_b8 two chunks ahead
      __builtin_prefetch(wp + (long long)(kc + 64) * N, 0, 1);
    __syncthreads();
    const v16bf af = load_a_frag(&aS[buf][mrow][0], 32);
    accs[0] = wmma_bf16(af, load_b_frag_t(&wS[buf][ncol][0], 32), accs[0]);
    accs[1] = wmma_bf16(af, load_b_frag_t(&wS[buf][ncol + 16][0], 32), accs[1]);
    buf ^= 1;
  }
  const int hf = lane >> 4, nlo = lane & 15;
  const bool clin = (csb == 0);
#pragma unroll
  for (int t = 0; t < 2; ++t) {
#pragma unroll
    for (int r = 0; r < 8; ++r) {
      const int row = m0 + mrow + r + hf * 8;
      const int col = n0 + ncol + t * 16 + nlo;
      if (col < N) {
        float v = accs[t][r];
        if (bias) v += bias[col];
        if (act == 1) v = v > 0.f ? v : 0.f;
        else if (act == 2) v = 1.f / (1.f + __expf(-v));
        if (mulbuf) v *= mulbuf[(long long)row * N + col];
        if (rowscale) v *= rowscale[row];
        const long long co = clin ? (long long)row * csl + col
                                  : (long long)(row / cL) * csb +
                                        (long long)(row % cL) * csl + col;
        float out = v;
        if (addC) out += C[co];
        C[co] = out;
      }
    }
  }
}

// ---------------------------------------------------------------------------
// Flash attention, dh=32, one wave per (b,h,16-query tile). Online softmax.
// bias[q,k,h] via strides; mask[b*L+k] additive.
// ---------------------------------------------------------------------------
__global__ __launch_bounds__(32) void attn_kernel(
    const float* __restrict__ Q, const float* __restrict__ Kx,
    const float* __restrict__ V, const float* __restrict__ mask,
    const float* __restrict__ bias, long long bq, long long bk, long long bh,
    float* __restrict__ O, int B, int L, int H, float scale) {
  __shared__ bf16 qS[16][32];
  __shared__ bf16 kS[32][32];   // [key][d]  (B-frag for Q*K^T: n=key,k=d)
  __shared__ bf16 vS[32][32];   // [d][key]  (B-frag for P*V  : n=d, k=key)
  __shared__ bf16 pS[16][32];
  const int lane = threadIdx.x & 31;
  const int QT = L >> 4;
  int bi = blockIdx.x;
  const int qt = bi % QT; bi /= QT;
  const int h = bi % H;   bi /= H;
  const int b = bi;
  const int q0 = qt * 16;
  const long long rs = (long long)H * 32;
  const float* qp = Q + (long long)b * L * rs + h * 32;
  const float* kp = Kx + (long long)b * L * rs + h * 32;
  const float* vp = V + (long long)b * L * rs + h * 32;
  for (int idx = lane; idx < 16 * 32; idx += 32) {
    const int r = idx >> 5, d = idx & 31;
    qS[r][d] = (bf16)(qp[(long long)(q0 + r) * rs + d] * scale);
  }
  WAIT_DS();
  const v16bf qf = load_a_frag(&qS[0][0], 32);
  const int hf = lane >> 4, nlo = lane & 15;
  float mI[8], lI[8];
  v8f acc0 = {}, acc1 = {};
#pragma unroll
  for (int r = 0; r < 8; ++r) { mI[r] = -1e30f; lI[r] = 0.f; }

  for (int kc = 0; kc < L; kc += 32) {
    for (int idx = lane; idx < 32 * 32; idx += 32) {
      const int kk = idx >> 5, d = idx & 31;
      const long long go = (long long)(kc + kk) * rs + d;
      kS[kk][d] = (bf16)kp[go];
      vS[d][kk] = (bf16)vp[go];
    }
    WAIT_DS();
    v8f z = {};
    v8f s0 = wmma_bf16(qf, load_b_frag_t(&kS[0][0], 32), z);
    v8f s1 = wmma_bf16(qf, load_b_frag_t(&kS[16][0], 32), z);
#pragma unroll
    for (int r = 0; r < 8; ++r) {
      const int qrow = q0 + r + hf * 8;
      const int kcol = kc + nlo;
      float x0 = s0[r], x1 = s1[r];
      if (bias) {
        const long long bb = (long long)qrow * bq + (long long)h * bh;
        x0 += bias[bb + (long long)kcol * bk];
        x1 += bias[bb + (long long)(kcol + 16) * bk];
      }
      if (mask) {
        x0 += mask[(long long)b * L + kcol];
        x1 += mask[(long long)b * L + kcol + 16];
      }
      float rm = fmaxf(x0, x1);
#pragma unroll
      for (int off = 8; off >= 1; off >>= 1) rm = fmaxf(rm, __shfl_xor(rm, off, 32));
      const float nm = fmaxf(mI[r], rm);
      const float alpha = __expf(mI[r] - nm);
      mI[r] = nm;
      const float p0 = __expf(x0 - nm), p1 = __expf(x1 - nm);
      float rsum = p0 + p1;
#pragma unroll
      for (int off = 8; off >= 1; off >>= 1) rsum += __shfl_xor(rsum, off, 32);
      lI[r] = lI[r] * alpha + rsum;
      acc0[r] *= alpha; acc1[r] *= alpha;
      pS[r + hf * 8][nlo] = (bf16)p0;
      pS[r + hf * 8][nlo + 16] = (bf16)p1;
    }
    WAIT_DS();
    const v16bf pf = load_a_frag(&pS[0][0], 32);
    acc0 = wmma_bf16(pf, load_b_frag_t(&vS[0][0], 32), acc0);
    acc1 = wmma_bf16(pf, load_b_frag_t(&vS[16][0], 32), acc1);
  }
  float* op = O + (long long)b * L * rs + h * 32;
#pragma unroll
  for (int r = 0; r < 8; ++r) {
    const int qrow = q0 + r + hf * 8;
    const float inv = lI[r] > 0.f ? 1.f / lI[r] : 0.f;
    op[(long long)qrow * rs + nlo] = acc0[r] * inv;
    op[(long long)qrow * rs + nlo + 16] = acc1[r] * inv;
  }
}

// ---------------------------------------------------------------------------
// Triangle-mult einsum: T[i,j,c] = sum_k a[i,k,c]b[j,k,c] (out) or a[k,i,c]b[k,j,c]
// grid (N/16, N/16, C/8); wave w handles channel c = bz*8+w.
// ---------------------------------------------------------------------------
__global__ __launch_bounds__(256) void trimul_kernel(
    const float* __restrict__ Aa, const float* __restrict__ Bb,
    float* __restrict__ T, int outgoing) {
  __shared__ bf16 aS[8][16][32];
  __shared__ bf16 bS[8][16][32];
  const int N = N_DIM, C = DPAIR;
  const int lane = threadIdx.x & 31, wave = threadIdx.x >> 5;
  const int i0 = blockIdx.x * 16, j0 = blockIdx.y * 16;
  const int c = blockIdx.z * 8 + wave;
  v8f acc = {};
  for (int kc = 0; kc < N; kc += 32) {
    for (int idx = lane; idx < 16 * 32; idx += 32) {
      const int r = idx >> 5, k = idx & 31;
      const long long ia = outgoing ? ((long long)(i0 + r) * N + (kc + k))
                                    : ((long long)(kc + k) * N + (i0 + r));
      const long long ib = outgoing ? ((long long)(j0 + r) * N + (kc + k))
                                    : ((long long)(kc + k) * N + (j0 + r));
      aS[wave][r][k] = (bf16)Aa[ia * C + c];
      bS[wave][r][k] = (bf16)Bb[ib * C + c];
    }
    WAIT_DS();
    acc = wmma_bf16(load_a_frag(&aS[wave][0][0], 32),
                    load_b_frag_t(&bS[wave][0][0], 32), acc);
  }
  const int hf = lane >> 4, nlo = lane & 15;
#pragma unroll
  for (int r = 0; r < 8; ++r)
    T[((long long)(i0 + r + hf * 8) * N + (j0 + nlo)) * C + c] = acc[r];
}

// ---------------------------------------------------------------------------
// Fused outer-product-mean: block = (i, 16-j tile).
// Phase1: O[j, c*32+d] = sum_s a[s,i,c] b[s,j,d]   (LDS, bf16)
// Phase2: z[i,j,:] += (O @ wo + bo) / (eps + norm[i,j])
// WoT is wo pre-transposed to bf16 [128][1024].
// ---------------------------------------------------------------------------
__global__ __launch_bounds__(256) void opm_kernel(
    const float* __restrict__ Aa, const float* __restrict__ Bb,
    const float* __restrict__ msk, const bf16* __restrict__ WoT,
    const float* __restrict__ bo, float* __restrict__ Z) {
  __shared__ bf16 aT[32][128];       // [c][s]
  __shared__ bf16 bT[8][32][128];    // per-wave [d][s]
  __shared__ bf16 OS[16][1024];      // [j][c*32+d]
  __shared__ float normS[16];
  const int tid = threadIdx.x, lane = tid & 31, wave = tid >> 5;
  const int hf = lane >> 4, nlo = lane & 15;
  const int i = blockIdx.x, j0 = blockIdx.y * 16;
  for (int idx = tid; idx < 32 * 128; idx += 256) {
    const int cc = idx >> 7, s = idx & 127;
    aT[cc][s] = (bf16)Aa[((long long)s * N_DIM + i) * 32 + cc];
  }
  if (tid < 16) {
    float nv = 0.f;
    for (int s = 0; s < S_DIM; ++s)
      nv += msk[s * N_DIM + i] * msk[s * N_DIM + j0 + tid];
    normS[tid] = nv;
  }
  __syncthreads();
  for (int jj = 0; jj < 2; ++jj) {
    const int j = wave + jj * 8;
    for (int idx = lane; idx < 128 * 32; idx += 32) {
      const int s = idx >> 5, d = idx & 31;
      bT[wave][d][s] = (bf16)Bb[((long long)s * N_DIM + (j0 + j)) * 32 + d];
    }
    WAIT_DS();
    v8f ac[2][2] = {{{}, {}}, {{}, {}}};
    for (int kc = 0; kc < 128; kc += 32) {
      v16bf a0 = load_a_frag(&aT[0][kc], 128);
      v16bf a1 = load_a_frag(&aT[16][kc], 128);
      v16bf b0 = load_b_frag_t(&bT[wave][0][kc], 128);
      v16bf b1 = load_b_frag_t(&bT[wave][16][kc], 128);
      ac[0][0] = wmma_bf16(a0, b0, ac[0][0]);
      ac[0][1] = wmma_bf16(a0, b1, ac[0][1]);
      ac[1][0] = wmma_bf16(a1, b0, ac[1][0]);
      ac[1][1] = wmma_bf16(a1, b1, ac[1][1]);
    }
#pragma unroll
    for (int mt = 0; mt < 2; ++mt)
#pragma unroll
      for (int nt = 0; nt < 2; ++nt)
#pragma unroll
        for (int r = 0; r < 8; ++r) {
          const int cr = mt * 16 + r + hf * 8;
          const int dc = nt * 16 + nlo;
          OS[j][cr * 32 + dc] = (bf16)ac[mt][nt][r];
        }
  }
  __syncthreads();
  v8f a2 = {};
  const int n0p = wave * 16;
  for (int kc = 0; kc < 1024; kc += 32) {
    a2 = wmma_bf16(load_a_frag(&OS[0][kc], 1024),
                   load_b_frag_gt(WoT, 1024, kc, n0p + nlo), a2);
  }
#pragma unroll
  for (int r = 0; r < 8; ++r) {
    const int jr = r + hf * 8;
    const int col = n0p + nlo;
    float v = a2[r] + bo[col];
    v /= (1e-3f + normS[jr]);
    const long long off = ((long long)i * N_DIM + (j0 + jr)) * DPAIR + col;
    Z[off] += v;
  }
}

// ---------------------------------------------------------------------------
// Fused transition MLP: Out += relu(X@W1+b1)@W2+b2 ; hidden lives in LDS.
// W1t bf16 [HN][D], W2t bf16 [D][HN] (pre-transposed). D,HN powers of two.
// ---------------------------------------------------------------------------
__global__ __launch_bounds__(256) void transition_kernel(
    const float* __restrict__ X, const bf16* __restrict__ W1t,
    const float* __restrict__ b1, const bf16* __restrict__ W2t,
    const float* __restrict__ b2, float* __restrict__ Out,
    long long osb, long long osl, int oL, int M, int D, int HN) {
  __shared__ bf16 xS[16 * 256];
  __shared__ bf16 hS[16 * 1024];
  const int tid = threadIdx.x, lane = tid & 31, wave = tid >> 5;
  const int hf = lane >> 4, nlo = lane & 15;
  const int m0 = blockIdx.x * 16;
  const int dsh = __builtin_ctz(D);
  for (int idx = tid; idx < 16 * D; idx += 256) {
    const int r = idx >> dsh, d = idx & (D - 1);
    xS[r * D + d] = (bf16)X[(long long)(m0 + r) * D + d];
  }
  __syncthreads();
  for (int nt = wave; nt < HN / 16; nt += 8) {
    v8f a = {};
    for (int kc = 0; kc < D; kc += 32)
      a = wmma_bf16(load_a_frag(&xS[kc], D),
                    load_b_frag_gt(W1t, D, kc, nt * 16 + nlo), a);
#pragma unroll
    for (int r = 0; r < 8; ++r) {
      const int row = r + hf * 8, col = nt * 16 + nlo;
      float v = a[r] + b1[col];
      hS[row * HN + col] = (bf16)(v > 0.f ? v : 0.f);
    }
  }
  __syncthreads();
  const bool olin = (osb == 0);
  for (int nt = wave; nt < D / 16; nt += 8) {
    v8f a = {};
    for (int kc = 0; kc < HN; kc += 32)
      a = wmma_bf16(load_a_frag(&hS[kc], HN),
                    load_b_frag_gt(W2t, HN, kc, nt * 16 + nlo), a);
#pragma unroll
    for (int r = 0; r < 8; ++r) {
      const int row = m0 + r + hf * 8, col = nt * 16 + nlo;
      const long long oo = olin ? (long long)row * osl + col
                                : (long long)(row / oL) * osb +
                                      (long long)(row % oL) * osl + col;
      Out[oo] += a[r] + b2[col];
    }
  }
}

// ---------------------------------------------------------------------------
// Host orchestration
// ---------------------------------------------------------------------------
extern "C" void kernel_launch(void* const* d_in, const int* in_sizes, int n_in,
                              void* d_out, int out_size, void* d_ws,
                              size_t ws_size, hipStream_t stream) {
  (void)in_sizes; (void)n_in; (void)out_size; (void)ws_size;
  const float* m_in       = (const float*)d_in[0];
  const float* z_in       = (const float*)d_in[1];
  const float* msa_mask   = (const float*)d_in[2];
  const float* pair_mask  = (const float*)d_in[3];
  const float* row_mask   = (const float*)d_in[4];
  const float* col_mask   = (const float*)d_in[5];
  const float* start_mask = (const float*)d_in[6];
  const float* end_mask   = (const float*)d_in[7];
  auto P = [&](int i) { return (const float*)d_in[8 + i]; };

  // params flattened in sorted-key (jax tree) order
  enum {
    MC = 0,   // msa_col:  bg bo ln_b ln_g wg wk wo wq wv                 (9)
    MR = 9,   // msa_row:  bg bo ln_m_b ln_m_g ln_z_b ln_z_g wb_pair wg wk wo wq wv (12)
    MT = 21,  // msa_trans: b1 b2 ln_b ln_g w1 w2                         (6)
    OP = 27,  // opm: ba bb bo ln_b ln_g wa wb wo                         (8)
    PT = 35,  // pair_trans: b1 b2 ln_b ln_g w1 w2                        (6)
    TE = 41,  // tri_att_end:  bg bo ln_b ln_g wb_pair wg wk wo wq wv     (10)
    TS = 51,  // tri_att_start                                            (10)
    TMI = 61, // tri_mul_in: ba bag bb bbg bg bo ln_in_b ln_in_g ln_out_b ln_out_g wa wag wb wbg wg wo (16)
    TMO = 77  // tri_mul_out                                              (16)
  };

  float* out_m = (float*)d_out;
  float* out_z = out_m + BIGF;
  float* W = (float*)d_ws;
  float* xln  = W + 0 * BIGF;
  float* qb   = W + 1 * BIGF;
  float* kb   = W + 2 * BIGF;
  float* vb   = W + 3 * BIGF;
  float* gate = W + 4 * BIGF;
  float* attb = W + 5 * BIGF;
  float* ta   = W + 6 * BIGF;
  float* tb   = W + 7 * BIGF;
  float* tt   = W + 8 * BIGF;
  float* biasb = W + 9 * BIGF;                    // up to N*N*HMSA floats
  bf16* wtp    = (bf16*)(W + 9 * BIGF + (1 << 20));
  bf16* mt_w1t = wtp;                 // [1024][256]
  bf16* mt_w2t = mt_w1t + 256 * 1024; // [256][1024]
  bf16* pt_w1t = mt_w2t + 256 * 1024; // [512][128]
  bf16* pt_w2t = pt_w1t + 128 * 512;  // [128][512]
  bf16* op_wot = pt_w2t + 128 * 512;  // [128][1024]

  hipMemcpyAsync(out_m, m_in, BIGF * sizeof(float), hipMemcpyDeviceToDevice, stream);
  hipMemcpyAsync(out_z, z_in, BIGF * sizeof(float), hipMemcpyDeviceToDevice, stream);

  // pre-transpose big weights to bf16 [n][k]
  transpose_bf16_kernel<<<dim3(256), dim3(256), 0, stream>>>(P(MT + 4), mt_w1t, 256, 10);
  transpose_bf16_kernel<<<dim3(256), dim3(256), 0, stream>>>(P(MT + 5), mt_w2t, 1024, 8);
  transpose_bf16_kernel<<<dim3(64), dim3(256), 0, stream>>>(P(PT + 4), pt_w1t, 128, 9);
  transpose_bf16_kernel<<<dim3(64), dim3(256), 0, stream>>>(P(PT + 5), pt_w2t, 512, 7);
  transpose_bf16_kernel<<<dim3(128), dim3(256), 0, stream>>>(P(OP + 7), op_wot, 1024, 7);

  const int Mm = S_DIM * N_DIM;   // 32768
  const int Mz = N_DIM * N_DIM;   // 65536
  const float scale = 0.1767766952966369f;  // 1/sqrt(32)

  auto LN = [&](const float* X, long long sb, long long sl, int L,
                const float* g, const float* b, float* Y, int M, int D) {
    ln_kernel<<<dim3((M + 7) / 8), dim3(256), 0, stream>>>(X, sb, sl, L, g, b, Y, M, D);
  };
  auto GEMM = [&](const float* A, long long asb, long long asl, int aL,
                  const float* Wt, const float* bias, float* C, long long csb,
                  long long csl, int cL, const float* mul, const float* rs,
                  int M, int K, int Nn, int act, int addC) {
    gemm_kernel<<<dim3((M + 63) / 64, (Nn + 63) / 64), dim3(256), 0, stream>>>(
        A, asb, asl, aL, Wt, bias, C, csb, csl, cL, mul, rs, M, K, Nn, act, addC);
  };
  auto ATTN = [&](const float* Q, const float* Kx, const float* V,
                  const float* mask, const float* bias, long long bq,
                  long long bk, long long bh, float* O, int B, int L, int H) {
    attn_kernel<<<dim3(B * H * (L / 16)), dim3(32), 0, stream>>>(
        Q, Kx, V, mask, bias, bq, bk, bh, O, B, L, H, scale);
  };

  // ===== MSA row attention with pair bias =====
  LN(out_m, 0, DMSA, Mm, P(MR + 3), P(MR + 2), xln, Mm, DMSA);
  LN(out_z, 0, DPAIR, Mz, P(MR + 5), P(MR + 4), ta, Mz, DPAIR);
  GEMM(ta, 0, DPAIR, Mz, P(MR + 6), nullptr, biasb, 0, HMSA, Mz, nullptr, nullptr,
       Mz, DPAIR, HMSA, 0, 0);
  GEMM(xln, 0, DMSA, Mm, P(MR + 10), nullptr, qb, 0, DMSA, Mm, nullptr, nullptr,
       Mm, DMSA, DMSA, 0, 0);
  GEMM(xln, 0, DMSA, Mm, P(MR + 8), nullptr, kb, 0, DMSA, Mm, nullptr, nullptr,
       Mm, DMSA, DMSA, 0, 0);
  GEMM(xln, 0, DMSA, Mm, P(MR + 11), nullptr, vb, 0, DMSA, Mm, nullptr, nullptr,
       Mm, DMSA, DMSA, 0, 0);
  ATTN(qb, kb, vb, row_mask, biasb, (long long)N_DIM * HMSA, HMSA, 1, attb,
       S_DIM, N_DIM, HMSA);
  GEMM(xln, 0, DMSA, Mm, P(MR + 7), P(MR + 0), gate, 0, DMSA, Mm, attb, nullptr,
       Mm, DMSA, DMSA, 2, 0);
  GEMM(gate, 0, DMSA, Mm, P(MR + 9), P(MR + 1), out_m, 0, DMSA, Mm, nullptr,
       nullptr, Mm, DMSA, DMSA, 0, 1);

  // ===== MSA column attention (logical transpose via strides) =====
  LN(out_m, DMSA, (long long)N_DIM * DMSA, S_DIM, P(MC + 3), P(MC + 2), xln, Mm, DMSA);
  GEMM(xln, 0, DMSA, Mm, P(MC + 7), nullptr, qb, 0, DMSA, Mm, nullptr, nullptr,
       Mm, DMSA, DMSA, 0, 0);
  GEMM(xln, 0, DMSA, Mm, P(MC + 5), nullptr, kb, 0, DMSA, Mm, nullptr, nullptr,
       Mm, DMSA, DMSA, 0, 0);
  GEMM(xln, 0, DMSA, Mm, P(MC + 8), nullptr, vb, 0, DMSA, Mm, nullptr, nullptr,
       Mm, DMSA, DMSA, 0, 0);
  ATTN(qb, kb, vb, col_mask, nullptr, 0, 0, 0, attb, N_DIM, S_DIM, HMSA);
  GEMM(xln, 0, DMSA, Mm, P(MC + 4), P(MC + 0), gate, 0, DMSA, Mm, attb, nullptr,
       Mm, DMSA, DMSA, 2, 0);
  GEMM(gate, 0, DMSA, Mm, P(MC + 6), P(MC + 1), out_m, DMSA,
       (long long)N_DIM * DMSA, S_DIM, nullptr, nullptr, Mm, DMSA, DMSA, 0, 1);

  // ===== MSA transition =====
  LN(out_m, 0, DMSA, Mm, P(MT + 3), P(MT + 2), xln, Mm, DMSA);
  transition_kernel<<<dim3(Mm / 16), dim3(256), 0, stream>>>(
      xln, mt_w1t, P(MT + 0), mt_w2t, P(MT + 1), out_m, 0, DMSA, Mm, Mm,
      DMSA, 4 * DMSA);

  // ===== Outer product mean =====
  LN(out_m, 0, DMSA, Mm, P(OP + 4), P(OP + 3), xln, Mm, DMSA);
  GEMM(xln, 0, DMSA, Mm, P(OP + 5), P(OP + 0), ta, 0, 32, Mm, nullptr, msa_mask,
       Mm, DMSA, 32, 0, 0);
  GEMM(xln, 0, DMSA, Mm, P(OP + 6), P(OP + 1), tb, 0, 32, Mm, nullptr, msa_mask,
       Mm, DMSA, 32, 0, 0);
  opm_kernel<<<dim3(N_DIM, N_DIM / 16), dim3(256), 0, stream>>>(
      ta, tb, msa_mask, op_wot, P(OP + 2), out_z);

  // ===== Triangle multiplicative updates =====
  auto TRIMUL = [&](int B, int outgoing) {
    LN(out_z, 0, DPAIR, Mz, P(B + 7), P(B + 6), xln, Mz, DPAIR);
    GEMM(xln, 0, DPAIR, Mz, P(B + 11), P(B + 1), gate, 0, DPAIR, Mz, nullptr,
         nullptr, Mz, DPAIR, DPAIR, 2, 0);
    GEMM(xln, 0, DPAIR, Mz, P(B + 10), P(B + 0), ta, 0, DPAIR, Mz, gate,
         pair_mask, Mz, DPAIR, DPAIR, 0, 0);
    GEMM(xln, 0, DPAIR, Mz, P(B + 13), P(B + 3), gate, 0, DPAIR, Mz, nullptr,
         nullptr, Mz, DPAIR, DPAIR, 2, 0);
    GEMM(xln, 0, DPAIR, Mz, P(B + 12), P(B + 2), tb, 0, DPAIR, Mz, gate,
         pair_mask, Mz, DPAIR, DPAIR, 0, 0);
    trimul_kernel<<<dim3(N_DIM / 16, N_DIM / 16, DPAIR / 8), dim3(256), 0,
                    stream>>>(ta, tb, tt, outgoing);
    LN(tt, 0, DPAIR, Mz, P(B + 9), P(B + 8), qb, Mz, DPAIR);
    GEMM(xln, 0, DPAIR, Mz, P(B + 14), P(B + 4), gate, 0, DPAIR, Mz, nullptr,
         nullptr, Mz, DPAIR, DPAIR, 2, 0);
    GEMM(qb, 0, DPAIR, Mz, P(B + 15), P(B + 5), out_z, 0, DPAIR, Mz, gate,
         nullptr, Mz, DPAIR, DPAIR, 0, 1);
  };
  TRIMUL(TMO, 1);
  TRIMUL(TMI, 0);

  // ===== Triangle attention (start / end) =====
  auto TRIATT = [&](int B, const float* mask, int transposed) {
    const long long sb = transposed ? DPAIR : 0;
    const long long sl = transposed ? (long long)N_DIM * DPAIR : DPAIR;
    const int L = transposed ? N_DIM : Mz;
    LN(out_z, sb, sl, L, P(B + 3), P(B + 2), xln, Mz, DPAIR);
    GEMM(xln, 0, DPAIR, Mz, P(B + 4), nullptr, biasb, 0, HPAIR, Mz, nullptr,
         nullptr, Mz, DPAIR, HPAIR, 0, 0);
    GEMM(xln, 0, DPAIR, Mz, P(B + 8), nullptr, qb, 0, DPAIR, Mz, nullptr,
         nullptr, Mz, DPAIR, DPAIR, 0, 0);
    GEMM(xln, 0, DPAIR, Mz, P(B + 6), nullptr, kb, 0, DPAIR, Mz, nullptr,
         nullptr, Mz, DPAIR, DPAIR, 0, 0);
    GEMM(xln, 0, DPAIR, Mz, P(B + 9), nullptr, vb, 0, DPAIR, Mz, nullptr,
         nullptr, Mz, DPAIR, DPAIR, 0, 0);
    ATTN(qb, kb, vb, mask, biasb, (long long)N_DIM * HPAIR, HPAIR, 1, attb,
         N_DIM, N_DIM, HPAIR);
    GEMM(xln, 0, DPAIR, Mz, P(B + 5), P(B + 0), gate, 0, DPAIR, Mz, attb,
         nullptr, Mz, DPAIR, DPAIR, 2, 0);
    GEMM(gate, 0, DPAIR, Mz, P(B + 7), P(B + 1), out_z, sb, sl, L, nullptr,
         nullptr, Mz, DPAIR, DPAIR, 0, 1);
  };
  TRIATT(TS, start_mask, 0);
  TRIATT(TE, end_mask, 1);

  // ===== Pair transition =====
  LN(out_z, 0, DPAIR, Mz, P(PT + 3), P(PT + 2), xln, Mz, DPAIR);
  transition_kernel<<<dim3(Mz / 16), dim3(256), 0, stream>>>(
      xln, pt_w1t, P(PT + 0), pt_w2t, P(PT + 1), out_z, 0, DPAIR, Mz, Mz,
      DPAIR, 4 * DPAIR);
}
